// OracleLocalRewriteModel_49160195670239
// MI455X (gfx1250) — matrix operand order repaired and checked by reading, not compiled, not run on hardware
//
#include <hip/hip_runtime.h>
#include <hip/hip_bf16.h>

typedef __attribute__((ext_vector_type(16))) _Float16 v16h;
typedef __attribute__((ext_vector_type(8)))  float    v8f;

#define BB 2
#define NN 320
#define FF 5
#define HH 128
#define MP_LAYERS 3
#define COPYV 10.0f
#define NEGV  -1e9f

// workspace layout (float offsets)
#define OFF_HA   0
#define OFF_HB   81920
#define OFF_DEG  163840
#define OFF_MSG  164480
#define OFF_P    246400
#define OFF_Q    328320
#define OFF_ERAW 410240
#define WS_FLOATS 615040   // bpack (f16) starts after this

// ---------------- degree: deg[b,i] = max(1, sum_j adj) ----------------
__global__ void k_deg(const float* __restrict__ adj, float* __restrict__ deg) {
  int bi = blockIdx.x;
  int lane = threadIdx.x;
  const float* row = adj + (size_t)bi * NN;
  float s = 0.f;
  for (int j = lane; j < NN; j += 32) s += row[j];
#pragma unroll
  for (int off = 16; off > 0; off >>= 1) s += __shfl_xor(s, off, 32);
  if (lane == 0) deg[bi] = fmaxf(s, 1.0f);
}

// ---------------- input projection: h = relu(nf @ W_in + b_in) ----------------
__global__ void k_inproj(const float* __restrict__ nf, const float* __restrict__ Win,
                         const float* __restrict__ bin, float* __restrict__ h) {
  int bi = blockIdx.x; int t = threadIdx.x;
  const float* x = nf + (size_t)bi * FF;
  float acc = bin[t];
#pragma unroll
  for (int k = 0; k < FF; ++k) acc += x[k] * Win[k*HH + t];
  h[(size_t)bi*HH + t] = fmaxf(acc, 0.f);
}

// ---------------- msg = (adj @ h) / deg ----------------
__global__ void k_msg(const float* __restrict__ adj, const float* __restrict__ h,
                      const float* __restrict__ deg, float* __restrict__ msg) {
  int bi = blockIdx.x; int t = threadIdx.x;
  int b = bi / NN;
  const float* arow = adj + (size_t)bi * NN;
  const float* hb = h + (size_t)b * NN * HH;
  float s = 0.f;
  for (int j = 0; j < NN; ++j) s += arow[j] * hb[(size_t)j*HH + t];
  msg[(size_t)bi*HH + t] = s / deg[bi];
}

// ---------------- h = relu(h@Wself + msg@Wmsg + b) ----------------
__global__ void k_mp(const float* __restrict__ h, const float* __restrict__ msg,
                     const float* __restrict__ Wself, const float* __restrict__ Wmsg,
                     const float* __restrict__ bmp, float* __restrict__ hout) {
  __shared__ float hr[HH];
  __shared__ float mr[HH];
  int bi = blockIdx.x; int t = threadIdx.x;
  hr[t] = h[(size_t)bi*HH + t];
  mr[t] = msg[(size_t)bi*HH + t];
  __syncthreads();
  float acc = bmp[t];
  for (int k = 0; k < HH; ++k) acc += hr[k]*Wself[k*HH+t] + mr[k]*Wmsg[k*HH+t];
  hout[(size_t)bi*HH + t] = fmaxf(acc, 0.f);
}

// ---------------- type/state heads + copy-merge, also emits h output ----------------
__global__ void k_heads(const float* __restrict__ h, const float* __restrict__ nf,
                        const float* __restrict__ snm,
                        const float* __restrict__ tW1, const float* __restrict__ tb1,
                        const float* __restrict__ tW2, const float* __restrict__ tb2,
                        const float* __restrict__ sW1, const float* __restrict__ sb1,
                        const float* __restrict__ sW2, const float* __restrict__ sb2,
                        float* __restrict__ out_h, float* __restrict__ out_tl,
                        float* __restrict__ out_sl, float* __restrict__ out_tf,
                        float* __restrict__ out_sf) {
  __shared__ float hr[HH];
  __shared__ float th[HH];
  __shared__ float sh[HH];
  int bi = blockIdx.x; int t = threadIdx.x;
  float hv = h[(size_t)bi*HH + t];
  hr[t] = hv;
  out_h[(size_t)bi*HH + t] = hv;
  __syncthreads();
  float a1 = tb1[t], a2 = sb1[t];
  for (int k = 0; k < HH; ++k) { float hk = hr[k]; a1 += hk*tW1[k*HH+t]; a2 += hk*sW1[k*HH+t]; }
  th[t] = fmaxf(a1, 0.f); sh[t] = fmaxf(a2, 0.f);
  __syncthreads();
  float scope = snm[bi];
  if (t < 3) {
    float v = tb2[t];
    for (int k = 0; k < HH; ++k) v += th[k]*tW2[k*3+t];
    out_tl[bi*3+t] = v;
    int ct = (int)nf[(size_t)bi*FF];
    ct = ct < 0 ? 0 : (ct > 2 ? 2 : ct);
    float cp = (t == ct) ? COPYV : -COPYV;
    out_tf[bi*3+t] = (scope != 0.f) ? v : cp;
  }
  if (t < 4) {
    float v = sb2[t];
    for (int k = 0; k < HH; ++k) v += sh[k]*sW2[k*4+t];
    out_sl[bi*4+t] = v;
    out_sf[bi*4+t] = scope*v + (1.f - scope)*nf[(size_t)bi*FF + 1 + t];
  }
}

// ---------------- per-node precompute: P = h@eW1[0:128] + eb1, Q = h@eW1[128:256] ----------------
__global__ void k_pq(const float* __restrict__ h, const float* __restrict__ eW1,
                     const float* __restrict__ eb1, float* __restrict__ P, float* __restrict__ Q) {
  __shared__ float hr[HH];
  int bi = blockIdx.x; int t = threadIdx.x;
  hr[t] = h[(size_t)bi*HH + t];
  __syncthreads();
  float p = eb1[t], q = 0.f;
  for (int k = 0; k < HH; ++k) {
    float hk = hr[k];
    p += hk * eW1[(size_t)k*HH + t];
    q += hk * eW1[(size_t)(128 + k)*HH + t];
  }
  P[(size_t)bi*HH + t] = p;
  Q[(size_t)bi*HH + t] = q;
}

// ---------------- pack eW1 rows 256..511 into WMMA B-operand layout (f16) ----------------
// Layout per (K=32) x 16(N) tile: lane n=lane&15 holds column n; lanes 0-15 K=0..7,16..23,
// lanes 16-31 K=8..15,24..31, 16 contiguous halfs per lane -> one v16h load in the hot loop.
__global__ void k_bpack(const float* __restrict__ eW1, _Float16* __restrict__ bp) {
  int blk = blockIdx.x;              // nt*8 + kb, 64 blocks
  int nt = blk >> 3, kb = blk & 7;
  int lane = threadIdx.x;
  int n = nt*16 + (lane & 15);
  int kbase = kb*32 + ((lane < 16) ? 0 : 8);
  _Float16* dst = bp + ((size_t)blk*32 + lane)*16;
#pragma unroll
  for (int e = 0; e < 16; ++e) {
    int K = kbase + ((e < 8) ? e : e + 8);  // 0..255 within pairwise K
    dst[e] = (_Float16)eW1[(size_t)(256 + K)*HH + n];
  }
}

// ---------------- pairwise edge MLP: the dominant GEMM, on the WMMA pipe ----------------
// wave tile: 16 pairs (i0..i0+15, fixed j) x 128 out channels; K=256 (|hi-hj| ++ hi*hj)
__global__ void __launch_bounds__(256) k_edge(const float* __restrict__ h,
                                              const float* __restrict__ P,
                                              const float* __restrict__ Q,
                                              const _Float16* __restrict__ bp,
                                              const float* __restrict__ eW2,
                                              const float* __restrict__ eb2,
                                              float* __restrict__ eraw) {
  __shared__ float hi_s[16*132];   // stride 132 -> conflict-free column reads
  __shared__ float hj_s[8*128];
  __shared__ float w2_s[128];

  int bid = blockIdx.x;            // 1600 = 2 * 20 * 40
  int b = bid / 800;
  int rem = bid % 800;
  int iblk = rem / 40, jgrp = rem % 40;
  int i0 = iblk * 16;
  int tid = threadIdx.x;
  int wv = tid >> 5, lane = tid & 31;

  for (int k = tid; k < 16*128; k += 256) {
    int row = k >> 7, col = k & 127;
    hi_s[row*132 + col] = h[((size_t)b*NN + i0 + row)*HH + col];
  }
  for (int k = tid; k < 8*128; k += 256) {
    int row = k >> 7, col = k & 127;
    hj_s[row*128 + col] = h[((size_t)b*NN + jgrp*8 + row)*HH + col];
  }
  if (tid < 128) w2_s[tid] = eW2[tid];
  __syncthreads();

  int j = jgrp*8 + wv;
  int m = lane & 15;               // A-matrix row for this lane
  int hi8 = lane >> 4;             // C-matrix M offset group (0 or 8)
  int kbase = hi8 * 8;             // A-matrix K split per ISA layout

  // Build A (16 pairs x 256 K) once; reused across all 8 N-tiles.
  // |x| expressed as fmaxf(d, -d) via two subtractions: avoids the broken
  // vector-fabs (bitcast+and) isel path in the gfx1250 backend.
  v16h A[8];
#pragma unroll
  for (int kb = 0; kb < 8; ++kb) {
    v16h av;
#pragma unroll
    for (int e = 0; e < 16; ++e) {
      int K = kb*32 + kbase + ((e < 8) ? e : e + 8);
      float val;
      if (K < 128) {
        float a = hi_s[m*132 + K];
        float c = hj_s[wv*128 + K];
        val = fmaxf(a - c, c - a);
      } else {
        int f = K - 128;
        val = hi_s[m*132 + f] * hj_s[wv*128 + f];
      }
      av[e] = (_Float16)val;
    }
    A[kb] = av;
  }

  float dot[8];
#pragma unroll
  for (int rr = 0; rr < 8; ++rr) dot[rr] = 0.f;

  const float* Prow = P + ((size_t)b*NN + i0 + hi8*8)*HH;
  const float* Qrow = Q + ((size_t)b*NN + j)*HH;

  for (int nt = 0; nt < 8; ++nt) {
    int n = nt*16 + (lane & 15);
    v8f acc;
#pragma unroll
    for (int rr = 0; rr < 8; ++rr) acc[rr] = Prow[(size_t)rr*HH + n] + Qrow[n];  // folds h_i,h_j GEMM terms
#pragma unroll
    for (int kb = 0; kb < 8; ++kb) {
      const v16h bv = *(const v16h*)(bp + (((size_t)nt*8 + kb)*32 + lane)*16);
      acc = __builtin_amdgcn_wmma_f32_16x16x32_f16(false, A[kb], false, bv,
                                                   (short)0, acc, false, false);
    }
    float w2 = w2_s[n];
#pragma unroll
    for (int rr = 0; rr < 8; ++rr) dot[rr] += fmaxf(acc[rr], 0.f) * w2;  // fused relu + eW2 dot
  }

  // reduce across the 16 lanes holding each M-row's N-slices
#pragma unroll
  for (int rr = 0; rr < 8; ++rr) {
#pragma unroll
    for (int off = 1; off < 16; off <<= 1)
      dot[rr] += __shfl_xor(dot[rr], off, 32);
  }
  if ((lane & 15) == 0) {
    float b2 = eb2[0];
#pragma unroll
    for (int rr = 0; rr < 8; ++rr)
      eraw[((size_t)b*NN + i0 + hi8*8 + rr)*NN + j] = dot[rr] + b2;
  }
}

// ---------------- symmetrize + diagonal NEG + copy/scope merge ----------------
__global__ void k_edgefin(const float* __restrict__ eraw, const float* __restrict__ adj,
                          const float* __restrict__ sem,
                          float* __restrict__ out_el, float* __restrict__ out_ef) {
  int idx = blockIdx.x*256 + threadIdx.x;
  if (idx >= BB*NN*NN) return;
  int b = idx / (NN*NN);
  int rem = idx % (NN*NN);
  int i = rem / NN, j = rem % NN;
  float el;
  if (i == j) el = NEGV;
  else el = 0.5f*(eraw[idx] + eraw[(size_t)b*NN*NN + (size_t)j*NN + i]);
  out_el[idx] = el;
  float ce = (i == j) ? NEGV : ((adj[idx] > 0.5f) ? COPYV : -COPYV);
  float ef = (sem[idx] != 0.f) ? el : ce;
  if (i == j) ef = NEGV;
  out_ef[idx] = ef;
}

extern "C" void kernel_launch(void* const* d_in, const int* in_sizes, int n_in,
                              void* d_out, int out_size, void* d_ws, size_t ws_size,
                              hipStream_t stream) {
  const float* nf    = (const float*)d_in[0];
  const float* adj   = (const float*)d_in[1];
  const float* snm   = (const float*)d_in[2];
  const float* sem   = (const float*)d_in[3];
  const float* Win   = (const float*)d_in[4];
  const float* bin   = (const float*)d_in[5];
  const float* Wself = (const float*)d_in[6];
  const float* Wmsg  = (const float*)d_in[7];
  const float* bmp   = (const float*)d_in[8];
  const float* tW1   = (const float*)d_in[9];
  const float* tb1   = (const float*)d_in[10];
  const float* tW2   = (const float*)d_in[11];
  const float* tb2   = (const float*)d_in[12];
  const float* sW1   = (const float*)d_in[13];
  const float* sb1   = (const float*)d_in[14];
  const float* sW2   = (const float*)d_in[15];
  const float* sb2   = (const float*)d_in[16];
  const float* eW1   = (const float*)d_in[17];
  const float* eb1   = (const float*)d_in[18];
  const float* eW2   = (const float*)d_in[19];
  const float* eb2   = (const float*)d_in[20];

  float* ws   = (float*)d_ws;
  float* hA   = ws + OFF_HA;
  float* hB   = ws + OFF_HB;
  float* deg  = ws + OFF_DEG;
  float* msg  = ws + OFF_MSG;
  float* P    = ws + OFF_P;
  float* Q    = ws + OFF_Q;
  float* eraw = ws + OFF_ERAW;
  _Float16* bp = (_Float16*)(ws + WS_FLOATS);

  float* out    = (float*)d_out;
  float* out_h  = out;              // [2,320,128]
  float* out_tl = out + 81920;      // [2,320,3]
  float* out_sl = out + 83840;      // [2,320,4]
  float* out_el = out + 86400;      // [2,320,320]
  float* out_tf = out + 291200;     // [2,320,3]
  float* out_sf = out + 293120;     // [2,320,4]
  float* out_ef = out + 295680;     // [2,320,320]

  k_deg<<<BB*NN, 32, 0, stream>>>(adj, deg);
  k_inproj<<<BB*NN, HH, 0, stream>>>(nf, Win, bin, hA);

  float* cur = hA; float* nxt = hB;
  for (int l = 0; l < MP_LAYERS; ++l) {
    k_msg<<<BB*NN, HH, 0, stream>>>(adj, cur, deg, msg);
    k_mp<<<BB*NN, HH, 0, stream>>>(cur, msg,
                                   Wself + (size_t)l*HH*HH, Wmsg + (size_t)l*HH*HH,
                                   bmp + (size_t)l*HH, nxt);
    float* t = cur; cur = nxt; nxt = t;
  }

  k_heads<<<BB*NN, HH, 0, stream>>>(cur, nf, snm, tW1, tb1, tW2, tb2,
                                    sW1, sb1, sW2, sb2,
                                    out_h, out_tl, out_sl, out_tf, out_sf);
  k_pq<<<BB*NN, HH, 0, stream>>>(cur, eW1, eb1, P, Q);
  k_bpack<<<64, 32, 0, stream>>>(eW1, bp);
  k_edge<<<BB*20*40, 256, 0, stream>>>(cur, P, Q, bp, eW2, eb2, eraw);
  k_edgefin<<<(BB*NN*NN + 255)/256, 256, 0, stream>>>(eraw, adj, sem, out_el, out_ef);
}
